// LSTMWithInputCellAttention_37606733644123
// MI455X (gfx1250) — matrix operand order repaired and checked by reading, not compiled, not run on hardware
//
#include <hip/hip_runtime.h>

typedef __attribute__((ext_vector_type(16))) __bf16 v16bf;
typedef __attribute__((ext_vector_type(8)))  float  v8f;

#define BS 64
#define TT 512
#define FF 128
#define HSZ 256
#define G4 1024
#define RR 8
#define DA 64

__device__ __forceinline__ float sigm(float x) { return 1.f / (1.f + __expf(-x)); }

__device__ __forceinline__ v8f wmma_bf16(v16bf a, v16bf b, v8f c) {
  // D = A(16x32 bf16) * B(32x16 bf16) + C(16x16 f32)
  return __builtin_amdgcn_wmma_f32_16x16x32_bf16(false, a, false, b, (short)0, c, false, false);
}

// Load a 16x32 bf16 tile (row-major, leading dim lda) into the WMMA A-operand
// lane layout (ISA 7.12.2): lane half h, row m=lane&15; VGPR v holds
// K = (v>>2)*16 + h*8 + (v&3)*2 and K+1.  Per lane this is two contiguous
// 16-byte runs -> compiler emits global_load_b128 / ds_load_b128 pairs.
__device__ __forceinline__ v16bf load_a_tile(const __bf16* A, int lda, int lane) {
  const int half = lane >> 4;
  const int m = lane & 15;
  v16bf a;
#pragma unroll
  for (int v = 0; v < 8; ++v) {
    const int k = ((v >> 2) << 4) + (half << 3) + ((v & 3) << 1);
    a[2 * v]     = A[m * lda + k];
    a[2 * v + 1] = A[m * lda + k + 1];
  }
  return a;
}

// ---------------------------------------------------------------------------
// Convert fp32 weight [K,N] to bf16, pre-swizzled into WMMA B-operand layout:
// tile (kt,nt) of 32x16; per lane (half=lane>>4, n=lane&15), element pair v is
// W[kt*32 + (v>>2)*16 + half*8 + (v&3)*2 .. +1][nt*16 + n].  Stored contiguously: 16 bf16/lane.
// ---------------------------------------------------------------------------
__global__ void prep_weights_kernel(const float* __restrict__ W, __bf16* __restrict__ out,
                                    int K, int N) {
  const int tid = blockIdx.x * blockDim.x + threadIdx.x;
  const int ntN = N >> 4;
  const int ntiles = (K >> 5) * ntN;
  if (tid >= ntiles * 32) return;
  const int tile = tid >> 5, lane = tid & 31;
  const int kt = tile / ntN, nt = tile - kt * ntN;
  const int half = lane >> 4, nl = lane & 15;
  const int n = nt * 16 + nl;
  __bf16* dst = out + (size_t)tid * 16;
#pragma unroll
  for (int v = 0; v < 8; ++v) {
    const int k = kt * 32 + ((v >> 2) << 4) + half * 8 + ((v & 3) << 1);
    dst[2 * v]     = (__bf16)W[(size_t)k * N + n];
    dst[2 * v + 1] = (__bf16)W[(size_t)(k + 1) * N + n];
  }
}

// ---------------------------------------------------------------------------
// scores[b,t,r] = tanh(x_row . W1[j,:] + b1[j]) . W2[r,:] + b2[r]
// One 64-thread block per (b,t) row.
// ---------------------------------------------------------------------------
__global__ void scores_kernel(const float* __restrict__ x, const float* __restrict__ w1,
                              const float* __restrict__ b1, const float* __restrict__ w2,
                              const float* __restrict__ b2, float* __restrict__ scores) {
  const int row = blockIdx.x;  // b*T + t
  const int tid = threadIdx.x; // 0..63
  __shared__ float xr[FF];
  __shared__ float hid[DA];
  xr[tid]      = x[(size_t)row * FF + tid];
  xr[tid + 64] = x[(size_t)row * FF + 64 + tid];
  __syncthreads();
  float acc = b1[tid];
#pragma unroll 4
  for (int k = 0; k < FF; ++k) acc += xr[k] * w1[tid * FF + k];
  hid[tid] = tanhf(acc);
  __syncthreads();
  if (tid < RR) {
    float s = b2[tid];
#pragma unroll 4
    for (int j = 0; j < DA; ++j) s += hid[j] * w2[tid * DA + j];
    scores[(size_t)row * RR + tid] = s;
  }
}

// ---------------------------------------------------------------------------
// Full-window softmax over t -> attention output [bs, r, T].
// One 256-thread block per (b, r).
// ---------------------------------------------------------------------------
__global__ void attn_kernel(const float* __restrict__ scores, float* __restrict__ att) {
  const int b = blockIdx.x >> 3;
  const int r = blockIdx.x & 7;
  const int tid = threadIdx.x; // 0..255
  __shared__ float red[256];
  const float s0 = scores[((size_t)b * TT + tid) * RR + r];
  const float s1 = scores[((size_t)b * TT + 256 + tid) * RR + r];
  red[tid] = fmaxf(s0, s1);
  __syncthreads();
  for (int s = 128; s > 0; s >>= 1) {
    if (tid < s) red[tid] = fmaxf(red[tid], red[tid + s]);
    __syncthreads();
  }
  const float mx = red[0];
  __syncthreads();
  const float e0 = __expf(s0 - mx), e1 = __expf(s1 - mx);
  red[tid] = e0 + e1;
  __syncthreads();
  for (int s = 128; s > 0; s >>= 1) {
    if (tid < s) red[tid] += red[tid + s];
    __syncthreads();
  }
  const float inv = 1.f / red[0];
  float* ap = att + ((size_t)b * RR + r) * TT;
  ap[tid]       = e0 * inv;
  ap[tid + 256] = e1 * inv;
}

// ---------------------------------------------------------------------------
// Online-softmax computation of M_t for all t; output bf16, t-major:
// Mb[(t*BS + b)*FF + f].  One 128-thread block per batch b, sequential in t.
// ---------------------------------------------------------------------------
__global__ void m_kernel(const float* __restrict__ x, const float* __restrict__ scores,
                         __bf16* __restrict__ Mb) {
  const int b = blockIdx.x;
  const int f = threadIdx.x; // 0..127
  float Nacc[RR], mx[RR], Z[RR];
#pragma unroll
  for (int r = 0; r < RR; ++r) { Nacc[r] = 0.f; Z[r] = 0.f; mx[r] = -INFINITY; }
  for (int t = 0; t < TT; ++t) {
    const float xf = x[((size_t)b * TT + t) * FF + f];
    const float* sp = scores + ((size_t)b * TT + t) * RR;
    float Msum = 0.f;
#pragma unroll
    for (int r = 0; r < RR; ++r) {
      const float s = sp[r];
      const float mn = fmaxf(mx[r], s);
      const float sc = __expf(mx[r] - mn);  // exp(-inf)=0 on first step
      const float e = __expf(s - mn);
      Z[r] = Z[r] * sc + e;
      Nacc[r] = Nacc[r] * sc + e * xf;
      mx[r] = mn;
      Msum += Nacc[r] / Z[r];
    }
    Mb[((size_t)t * BS + b) * FF + f] = (__bf16)(Msum * 0.125f);
  }
}

// ---------------------------------------------------------------------------
// gatesX = M_all(bf16) @ W_ih(bf16, swizzled) + bias.
// Output kept in WMMA C/D register layout, tile-major:
//   gatesX[((mt*64 + nt)*32 + lane)*8 .. +7]  (one v8f per lane per tile)
// so both the store here and the C-load in the recurrent kernel are single
// 32B-per-lane vector accesses (b128 pairs) instead of 8 scattered b32s.
// grid (2048, 8), 256 threads (8 waves): wave = one 16x16 tile, K loop = 4 WMMAs.
// ---------------------------------------------------------------------------
__global__ void gatesx_kernel(const __bf16* __restrict__ Mb, const __bf16* __restrict__ Wsw,
                              const float* __restrict__ bias, float* __restrict__ gatesX) {
  const int lane = threadIdx.x & 31;
  const int wave = threadIdx.x >> 5;
  const int mt = blockIdx.x;            // 0..2047 (t-major row tiles)
  const int nt = blockIdx.y * 8 + wave; // 0..63
  const int row0 = mt * 16;
  const float cb = bias[nt * 16 + (lane & 15)];
  v8f acc = {cb, cb, cb, cb, cb, cb, cb, cb};
#pragma unroll
  for (int kt = 0; kt < 4; ++kt) {
    v16bf a = load_a_tile(Mb + (size_t)row0 * FF + kt * 32, FF, lane);
    v16bf bt = *(const v16bf*)(Wsw + ((size_t)(kt * 64 + nt) * 32 + lane) * 16);
    acc = wmma_bf16(a, bt, acc);
  }
  *(v8f*)(gatesX + ((size_t)(mt * 64 + nt) * 32 + lane) * 8) = acc;
}

// ---------------------------------------------------------------------------
// Persistent recurrent kernel: 1 block, 1024 threads (32 waves).
// Per step: gates = gatesX[t] (vector-loaded in D layout) + h_{t-1} @ W_hh
// (WMMA bf16), LSTM elementwise, h double-buffered in LDS (2 x 32KB).
// Wave w: M-tile = w&3 (16 batch rows), N-group = w>>2 handles 2 HS column
// tiles x 4 gate blocks.
// ---------------------------------------------------------------------------
__global__ void __launch_bounds__(1024, 1)
recurrent_kernel(const float* __restrict__ gatesX, const __bf16* __restrict__ Whh_sw,
                 float* __restrict__ hidden_seq, float* __restrict__ h_out,
                 float* __restrict__ c_out) {
  __shared__ __bf16 hbuf[2][BS * HSZ]; // 2 x 32KB double buffer
  const int tid = threadIdx.x;
  const int lane = tid & 31, wave = tid >> 5;
  const int half = lane >> 4, nl = lane & 15;
  const int b0 = (wave & 3) * 16;
  const int nGroup = wave >> 2; // 0..7
  float c[2][8], hfin[2][8];
#pragma unroll
  for (int s = 0; s < 2; ++s)
#pragma unroll
    for (int v = 0; v < 8; ++v) { c[s][v] = 0.f; hfin[s][v] = 0.f; }
  for (int i = tid; i < BS * HSZ; i += 1024) hbuf[0][i] = (__bf16)0.f;
  __syncthreads();

  for (int t = 0; t < TT; ++t) {
    const __bf16* hr = hbuf[t & 1];
    __bf16* hw = hbuf[(t + 1) & 1];
    const int mt = t * 4 + (wave & 3); // this wave's row tile at step t
#pragma unroll
    for (int s = 0; s < 2; ++s) {
      const int nTile = nGroup * 2 + s;
      const int ncol0 = nTile * 16;
      v8f acc[4];
#pragma unroll
      for (int g = 0; g < 4; ++g) // C init: vector load in D layout (32B/lane)
        acc[g] = *(const v8f*)(gatesX +
                               ((size_t)(mt * 64 + g * 16 + nTile) * 32 + lane) * 8);
#pragma unroll
      for (int kt = 0; kt < 8; ++kt) {
        v16bf a = load_a_tile(hr + b0 * HSZ + kt * 32, HSZ, lane);
#pragma unroll
        for (int g = 0; g < 4; ++g) {
          v16bf bt = *(const v16bf*)(Whh_sw + ((size_t)(kt * 64 + g * 16 + nTile) * 32 + lane) * 16);
          acc[g] = wmma_bf16(a, bt, acc[g]);
        }
      }
#pragma unroll
      for (int v = 0; v < 8; ++v) {
        const float ig = sigm(acc[0][v]);
        const float fg = sigm(acc[1][v]);
        const float gg = tanhf(acc[2][v]);
        const float og = sigm(acc[3][v]);
        const float cn = fg * c[s][v] + ig * gg;
        c[s][v] = cn;
        const float hn = og * tanhf(cn);
        hfin[s][v] = hn;
        const int brow = b0 + v + 8 * half;
        const int ncol = ncol0 + nl;
        hw[brow * HSZ + ncol] = (__bf16)hn;
        hidden_seq[((size_t)brow * TT + t) * HSZ + ncol] = hn;
      }
    }
    __syncthreads(); // hw fully written before it becomes next step's hr
  }
#pragma unroll
  for (int s = 0; s < 2; ++s) {
    const int ncol0 = (nGroup * 2 + s) * 16;
#pragma unroll
    for (int v = 0; v < 8; ++v) {
      const int brow = b0 + v + 8 * half;
      const int ncol = ncol0 + nl;
      h_out[(size_t)brow * HSZ + ncol] = hfin[s][v];
      c_out[(size_t)brow * HSZ + ncol] = c[s][v];
    }
  }
}

extern "C" void kernel_launch(void* const* d_in, const int* in_sizes, int n_in,
                              void* d_out, int out_size, void* d_ws, size_t ws_size,
                              hipStream_t stream) {
  (void)in_sizes; (void)n_in; (void)out_size; (void)ws_size;
  const float* x       = (const float*)d_in[0];
  const float* w_first = (const float*)d_in[1];
  const float* b_first = (const float*)d_in[2];
  const float* w_second= (const float*)d_in[3];
  const float* b_second= (const float*)d_in[4];
  const float* w_iBarh = (const float*)d_in[5];
  const float* w_hh    = (const float*)d_in[6];
  const float* bias    = (const float*)d_in[7];

  float* out = (float*)d_out;
  float* hidden_seq = out;                                    // [64,512,256]
  float* h_t        = out + (size_t)BS * TT * HSZ;            // [64,256]
  float* c_t        = h_t + (size_t)BS * HSZ;                 // [64,256]
  float* attention  = c_t + (size_t)BS * HSZ;                 // [64,8,512]

  char* ws = (char*)d_ws;
  __bf16* wi_sw  = (__bf16*)(ws);                       // 256 KB (4*64 tiles)
  __bf16* whh_sw = (__bf16*)(ws + (256u << 10));        // 512 KB (8*64 tiles)
  float*  scores = (float*) (ws + (768u << 10));        // 1 MB [bs,T,r]
  __bf16* Mb     = (__bf16*)(ws + (1792u << 10));       // 8 MB  [t-major, bf16]
  float*  gatesX = (float*) (ws + (9984u << 10));       // 128 MB [tile-swizzled fp32]

  // 1) bf16-convert + WMMA-swizzle weights (L2 resident thereafter)
  prep_weights_kernel<<<(4 * 64 * 32 + 255) / 256, 256, 0, stream>>>(w_iBarh, wi_sw, FF, G4);
  prep_weights_kernel<<<(8 * 64 * 32 + 255) / 256, 256, 0, stream>>>(w_hh, whh_sw, HSZ, G4);
  // 2) attention scores (tiny)
  scores_kernel<<<BS * TT, 64, 0, stream>>>(x, w_first, b_first, w_second, b_second, scores);
  // 3) full-window softmax -> attention output
  attn_kernel<<<BS * RR, 256, 0, stream>>>(scores, attention);
  // 4) online-softmax M_t for all t (O(T) instead of O(T^2))
  m_kernel<<<BS, FF, 0, stream>>>(x, scores, Mb);
  // 5) gatesX = M @ W_ih + bias : 131072 WMMA tiles across the device
  gatesx_kernel<<<dim3(2048, 8), 256, 0, stream>>>(Mb, wi_sw, bias, gatesX);
  // 6) sequential recurrence on one persistent workgroup
  recurrent_kernel<<<1, 1024, 0, stream>>>(gatesX, whh_sw, hidden_seq, h_t, c_t);
}
